// EchoStateNetwork_11158325035091
// MI455X (gfx1250) — compile-verified, matrix operand order checked by reading, not run on hardware
//
#include <hip/hip_runtime.h>

#define N_    512
#define T_    1024
#define DIN_  64
#define R_    2048
#define DOUT_ 64
#define ALPHA_ 0.3f
#define LN_EPS_ 1e-5f

typedef __attribute__((ext_vector_type(16))) __bf16 v16bf;
typedef __attribute__((ext_vector_type(8)))  float v8f;
typedef __attribute__((ext_vector_type(4)))  int   v4i;

// pointee-qualified pointer types for the async DMA builtin
typedef __attribute__((address_space(1))) v4i* gptr_v4i;
typedef __attribute__((address_space(3))) v4i* lptr_v4i;

union FragBF { uint4 u[2]; v16bf v; };

// ---- CDNA5 async global->LDS path (probe-guarded) ----
#if defined(__has_builtin)
#  if __has_builtin(__builtin_amdgcn_global_load_async_to_lds_b128)
#    define ASYNC_LDS 1
#  endif
#endif
#ifndef ASYNC_LDS
#  define ASYNC_LDS 0
#endif

__device__ __forceinline__ unsigned short f2bf(float f) {
  union { float f; unsigned int u; } c; c.f = f;
  unsigned int u = c.u;
  u += 0x7FFFu + ((u >> 16) & 1u);          // round-to-nearest-even
  return (unsigned short)(u >> 16);
}
__device__ __forceinline__ float bf2f(unsigned short b) {
  union { unsigned int u; float f; } c; c.u = ((unsigned int)b) << 16;
  return c.f;
}

// Stage 32B (16 bf16) from global into LDS.
// Async path: two GLOBAL_LOAD_ASYNC_TO_LDS_B128 (ASYNCcnt, no VGPR staging).
__device__ __forceinline__ void stage32B(unsigned short* ldst,
                                         const unsigned short* gsrc) {
#if ASYNC_LDS
  gptr_v4i g0 = (gptr_v4i)gsrc;
  lptr_v4i l0 = (lptr_v4i)ldst;
  __builtin_amdgcn_global_load_async_to_lds_b128(g0,     l0,     0, 0);
  __builtin_amdgcn_global_load_async_to_lds_b128(g0 + 1, l0 + 1, 0, 0);
#else
  const uint4* sp = (const uint4*)gsrc;
  uint4* dp = (uint4*)ldst;
  dp[0] = sp[0]; dp[1] = sp[1];
#endif
}

__device__ __forceinline__ void wait_async_all() {
#if ASYNC_LDS
#  if __has_builtin(__builtin_amdgcn_s_wait_asynccnt)
  __builtin_amdgcn_s_wait_asynccnt(0);
#  else
  asm volatile("s_wait_asynccnt 0" ::: "memory");
#  endif
#endif
}

// ---------------- prep kernels ----------------
__global__ void cvt_bf16_kernel(const float* __restrict__ src,
                                unsigned short* __restrict__ dst, int n) {
  int i = blockIdx.x * blockDim.x + threadIdx.x;
  if (i < n) dst[i] = f2bf(src[i]);
}

__global__ void zero_u16_kernel(unsigned short* __restrict__ dst, int n) {
  int i = blockIdx.x * blockDim.x + threadIdx.x;
  if (i < n) dst[i] = 0;
}

// ---------------- per-timestep fused ESN step ----------------
// acc = x_t @ W_in^T + h @ W_res^T      (bf16 WMMA, f32 accumulate)
// h'  = (1-a)*h + a*tanh(acc)
// Block tile 64 (rows n) x 128 (cols r), 128 threads = 4 wave32 waves.
// Each wave owns a 32x64 tile: 2 M-frags x 4 N-frags = 8 accumulators,
// so every B fragment load feeds 2 WMMAs (1.5 ds_loads per WMMA).
// Double-buffered LDS stages; async DMA overlapped with WMMA compute.
__global__ void __launch_bounds__(128)
esn_step_kernel(const float* __restrict__ x,              // [N,T,DIN] f32
                const unsigned short* __restrict__ Win,   // [R,DIN]  bf16
                const unsigned short* __restrict__ Wres,  // [R,R]    bf16
                const unsigned short* __restrict__ h_src, // [N,R]    bf16
                unsigned short* __restrict__ h_dst,       // [N,R]    bf16
                int t) {
  __shared__ unsigned short As[2][64][72];    // h / x_t tiles (padded, 16B aligned)
  __shared__ unsigned short Bs[2][128][72];   // W row tiles (contiguous K)

  const int tid  = threadIdx.x;
  const int lane = tid & 31;
  const int wave = tid >> 5;     // 0..3
  const int wm   = wave & 1;     // 32-row half
  const int wn   = wave >> 1;    // 64-col half
  const int lr   = lane & 15;    // row (A) / col (B) within 16-tile
  const int khi  = lane >> 4;    // K phase: lanes 16-31 hold K 8-15 / 24-31

  const int nb = blockIdx.y * 64;    // batch-row base
  const int rb = blockIdx.x * 128;   // reservoir-col base

  v8f acc[2][4] = {};

  // 2 k-steps; per k-step: 2 A frags + 4 B frags feed 8 WMMAs
  auto compute = [&](int b) {
#pragma unroll
    for (int kk = 0; kk < 64; kk += 32) {
      const int ak = kk + khi * 8;
      FragBF a[2];
#pragma unroll
      for (int i = 0; i < 2; ++i) {
        const unsigned short* ar = &As[b][wm * 32 + i * 16 + lr][0];
        a[i].u[0] = *(const uint4*)(ar + ak);
        a[i].u[1] = *(const uint4*)(ar + ak + 16);
      }
#pragma unroll
      for (int j = 0; j < 4; ++j) {
        FragBF bb;
        const unsigned short* br = &Bs[b][wn * 64 + j * 16 + lr][0];
        bb.u[0] = *(const uint4*)(br + ak);
        bb.u[1] = *(const uint4*)(br + ak + 16);
#pragma unroll
        for (int i = 0; i < 2; ++i) {
          acc[i][j] = __builtin_amdgcn_wmma_f32_16x16x32_bf16(
              false, a[i].v, false, bb.v, (short)0, acc[i][j], false, false);
        }
      }
    }
  };

  // issue one recurrent K-stage (64 wide) into buffer b
  auto stage_rec = [&](int b, int ks) {
#pragma unroll
    for (int p = 0; p < 2; ++p) {      // A: 64 rows x 4 segs = 256 units
      int u = tid + p * 128;
      int row = u >> 2, seg = u & 3;
      stage32B(&As[b][row][seg * 16],
               h_src + (size_t)(nb + row) * R_ + ks + seg * 16);
    }
#pragma unroll
    for (int p = 0; p < 4; ++p) {      // B: 128 rows x 4 segs = 512 units
      int u = tid + p * 128;
      int row = u >> 2, seg = u & 3;
      stage32B(&Bs[b][row][seg * 16],
               Wres + (size_t)(rb + row) * R_ + ks + seg * 16);
    }
  };

  // ---- phase 1 staging into buffer 0: x_t (convert f32->bf16) + Win rows ----
  {
#pragma unroll
    for (int p = 0; p < 2; ++p) {
      int u = tid + p * 128;
      int row = u >> 2, seg = u & 3;
      const float4* xp =
          (const float4*)(x + ((size_t)(nb + row) * T_ + t) * DIN_ + seg * 16);
#pragma unroll
      for (int q = 0; q < 4; ++q) {
        float4 f = xp[q];
        int c = seg * 16 + q * 4;
        As[0][row][c + 0] = f2bf(f.x); As[0][row][c + 1] = f2bf(f.y);
        As[0][row][c + 2] = f2bf(f.z); As[0][row][c + 3] = f2bf(f.w);
      }
    }
#pragma unroll
    for (int p = 0; p < 4; ++p) {
      int u = tid + p * 128;
      int row = u >> 2, seg = u & 3;
      stage32B(&Bs[0][row][seg * 16],
               Win + (size_t)(rb + row) * DIN_ + seg * 16);
    }
  }
  wait_async_all();
  __syncthreads();

  // prefetch first recurrent stage while doing the input-projection WMMAs
  stage_rec(1, 0);
  compute(0);                 // u = x_t @ Win^T  (K=64)
  wait_async_all();
  __syncthreads();

  // ---- phase 2: recurrence, K = R = 2048 in double-buffered stages of 64 ----
  int buf = 1;
  for (int ks = 0; ks < R_; ks += 64) {
    if (ks + 64 < R_) stage_rec(buf ^ 1, ks + 64);   // DMA next stage
    compute(buf);                                    // WMMA current stage
    wait_async_all();
    __syncthreads();
    buf ^= 1;
  }

  // ---- epilogue: leaky-integrated tanh update ----
  // C/D layout: vgpr e, lanes 0-15 -> M=e, N=lane; lanes 16-31 -> M=8+e, N=lane-16
#pragma unroll
  for (int i = 0; i < 2; ++i) {
#pragma unroll
    for (int j = 0; j < 4; ++j) {
#pragma unroll
      for (int e = 0; e < 8; ++e) {
        int m   = wm * 32 + i * 16 + khi * 8 + e;
        int col = rb + wn * 64 + j * 16 + lr;
        size_t idx = (size_t)(nb + m) * R_ + col;
        float hold = bf2f(h_src[idx]);
        float v = (1.0f - ALPHA_) * hold + ALPHA_ * tanhf(acc[i][j][e]);
        h_dst[idx] = f2bf(v);
      }
    }
  }
}

// ---------------- readout (Linear) + LayerNorm ----------------
__global__ void __launch_bounds__(64)
readout_ln_kernel(const unsigned short* __restrict__ h,  // [N,R] bf16
                  const float* __restrict__ w_ro,        // [DOUT,R]
                  const float* __restrict__ b_ro,
                  const float* __restrict__ gamma,
                  const float* __restrict__ beta,
                  float* __restrict__ out) {             // [N,DOUT]
  __shared__ float hrow[R_];
  __shared__ float sy[DOUT_];
  const int n = blockIdx.x, tid = threadIdx.x;

  for (int i = tid; i < R_; i += 64) hrow[i] = bf2f(h[(size_t)n * R_ + i]);
  __syncthreads();

  float acc = 0.f;
  const float* w = w_ro + (size_t)tid * R_;
  for (int k = 0; k < R_; ++k) acc = fmaf(hrow[k], w[k], acc);
  float y = acc + b_ro[tid];
  sy[tid] = y;
  __syncthreads();

  float mu = 0.f;
  for (int i = 0; i < DOUT_; ++i) mu += sy[i];
  mu *= (1.0f / DOUT_);
  float var = 0.f;
  for (int i = 0; i < DOUT_; ++i) { float d = sy[i] - mu; var += d * d; }
  var *= (1.0f / DOUT_);

  out[(size_t)n * DOUT_ + tid] = (y - mu) * rsqrtf(var + LN_EPS_) * gamma[tid] + beta[tid];
}

// ---------------- host driver ----------------
extern "C" void kernel_launch(void* const* d_in, const int* in_sizes, int n_in,
                              void* d_out, int out_size, void* d_ws, size_t ws_size,
                              hipStream_t stream) {
  const float* x     = (const float*)d_in[0];
  const float* W_in  = (const float*)d_in[1];
  const float* W_res = (const float*)d_in[2];
  const float* w_ro  = (const float*)d_in[3];
  const float* b_ro  = (const float*)d_in[4];
  const float* gamma = (const float*)d_in[5];
  const float* beta  = (const float*)d_in[6];
  float* out = (float*)d_out;

  // scratch layout (bf16 words): Wres(8MB) | Win(256KB) | hA(2MB) | hB(2MB)
  unsigned short* Wres_bf = (unsigned short*)d_ws;
  unsigned short* Win_bf  = Wres_bf + (size_t)R_ * R_;
  unsigned short* hA      = Win_bf  + (size_t)R_ * DIN_;
  unsigned short* hB      = hA      + (size_t)N_ * R_;

  { int n = R_ * R_;   cvt_bf16_kernel<<<(n + 255) / 256, 256, 0, stream>>>(W_res, Wres_bf, n); }
  { int n = R_ * DIN_; cvt_bf16_kernel<<<(n + 255) / 256, 256, 0, stream>>>(W_in,  Win_bf,  n); }
  { int n = N_ * R_;   zero_u16_kernel<<<(n + 255) / 256, 256, 0, stream>>>(hA, n); }

  unsigned short* hsrc = hA;
  unsigned short* hdst = hB;
  dim3 grid(R_ / 128, N_ / 64);
  for (int t = 0; t < T_; ++t) {
    esn_step_kernel<<<grid, 128, 0, stream>>>(x, Win_bf, Wres_bf, hsrc, hdst, t);
    unsigned short* tmp = hsrc; hsrc = hdst; hdst = tmp;
  }

  readout_ln_kernel<<<N_, 64, 0, stream>>>(hsrc, w_ro, b_ro, gamma, beta, out);
}
